// GCN_65575560675752
// MI455X (gfx1250) — matrix operand order repaired and checked by reading, not compiled, not run on hardware
//
#include <hip/hip_runtime.h>
#include <math.h>

typedef __attribute__((ext_vector_type(16))) _Float16 v16h;
typedef __attribute__((ext_vector_type(8)))  float    v8f;

#define BN_EPS 1e-5f

// ---------------------------------------------------------------------------
// WMMA fragment helpers (CDNA5 16x16x32 f16, wave32)
// A layout (16-bit, 16x32): lanes 0-15 -> M=lane, K 0..15;
//                           lanes 16-31 -> M=lane-16, K 16..31.
// element i: k = 16*h + i (matches B)
// B layout (32x16): lane -> N=lane&15, h=lane>>4; element i -> K = 16*h + i
// C/D layout: lane -> N=lane&15, h=lane>>4; VGPR v -> M = v + 8*h
// B fragments are pre-swizzled into LDS as v16h[frag][lane] so the inner loop
// issues ds_load_b128 pairs instead of 16 scalar ds_load_u16.
// ---------------------------------------------------------------------------
__device__ __forceinline__ int a16_k(int i, int h) {
    return (h << 4) + i;
}

__device__ __forceinline__ v8f wmma_f16(v16h a, v16h b, v8f c) {
    return __builtin_amdgcn_wmma_f32_16x16x32_f16(false, a, false, b, (short)0, c, false, false);
}

// ---------------------------------------------------------------------------
// Utility kernels
// ---------------------------------------------------------------------------
__global__ void k_zero(float* __restrict__ p, long long n) {
    long long i = (long long)blockIdx.x * blockDim.x + threadIdx.x;
    if (i < n) p[i] = 0.0f;
}

__global__ void k_deg(const int* __restrict__ src, const int* __restrict__ dst,
                      float* __restrict__ deg_out, float* __restrict__ deg_in, int E) {
    int e = blockIdx.x * blockDim.x + threadIdx.x;
    if (e < E) {
        atomicAdd(&deg_out[src[e]], 1.0f);
        atomicAdd(&deg_in[dst[e]], 1.0f);
    }
}

// scatter-add: agg[dst[e]*F + f] += xs[src[e]*F + f]
template <int F>
__global__ void k_scat(const int* __restrict__ src, const int* __restrict__ dst,
                       const float* __restrict__ xs, float* __restrict__ agg, int E) {
    long long t = (long long)blockIdx.x * blockDim.x + threadIdx.x;
    if (t >= (long long)E * F) return;
    int e = (int)(t >> __builtin_ctz(F));
    int f = (int)(t & (F - 1));
    atomicAdd(&agg[(size_t)dst[e] * F + f], xs[(size_t)src[e] * F + f]);
}

// ---------------------------------------------------------------------------
// GEMM1: xs1[n,16] = (n_feats[n,:] @ W1) * rsqrt(max(deg_out,1))   (K=128)
// ---------------------------------------------------------------------------
__global__ void k_gemm1(const float* __restrict__ nf, const float* __restrict__ W1,
                        const float* __restrict__ deg_out, float* __restrict__ xs1, int N) {
    __shared__ v16h lB[4 * 32];                 // [kk][lane], pre-swizzled fragments
    {
        _Float16* pb = (_Float16*)lB;
        for (int idx = threadIdx.x; idx < 4 * 32 * 16; idx += blockDim.x) {
            int kk = idx >> 9, L = (idx >> 4) & 31, i = idx & 15;
            int k = kk * 32 + ((L >> 4) << 4) + i;      // K row for this element
            pb[idx] = (_Float16)W1[k * 16 + (L & 15)];
        }
    }
    __syncthreads();

    int wv = threadIdx.x >> 5, lane = threadIdx.x & 31;
    int tile = blockIdx.x * (blockDim.x >> 5) + wv;
    int m0 = tile * 16;
    if (m0 >= N) return;
    int h = lane >> 4, col = lane & 15;
    int row = m0 + (lane & 15); if (row >= N) row = N - 1;
    const float* ar = nf + (size_t)row * 128;

    v8f c = {};
#pragma unroll
    for (int kk = 0; kk < 4; ++kk) {
        v16h a;
#pragma unroll
        for (int i = 0; i < 16; ++i) a[i] = (_Float16)ar[kk * 32 + a16_k(i, h)];
        c = wmma_f16(a, lB[kk * 32 + lane], c);
    }
#pragma unroll
    for (int v = 0; v < 8; ++v) {
        int r = m0 + v + (h << 3);
        if (r < N) {
            float rs = rsqrtf(fmaxf(deg_out[r], 1.0f));
            xs1[(size_t)r * 16 + col] = c[v] * rs;
        }
    }
}

// finalize GC1: h1 = relu(agg1*rsqrt(max(deg_in,1)) + b1); accumulate BN sums
__global__ void k_fin1(const float* __restrict__ agg1, const float* __restrict__ deg_in,
                       const float* __restrict__ b1, float* __restrict__ h1,
                       float* __restrict__ stats, int N) {
    __shared__ float lsum[16], lsq[16];
    if (threadIdx.x < 16) { lsum[threadIdx.x] = 0.f; lsq[threadIdx.x] = 0.f; }
    __syncthreads();
    long long idx = (long long)blockIdx.x * blockDim.x + threadIdx.x;
    if (idx < (long long)N * 16) {
        int nd = (int)(idx >> 4), f = (int)(idx & 15);
        float v = agg1[idx] * rsqrtf(fmaxf(deg_in[nd], 1.0f)) + b1[f];
        v = fmaxf(v, 0.0f);
        h1[idx] = v;
        atomicAdd(&lsum[f], v);
        atomicAdd(&lsq[f], v * v);
    }
    __syncthreads();
    if (threadIdx.x < 16) {
        atomicAdd(&stats[threadIdx.x], lsum[threadIdx.x]);
        atomicAdd(&stats[16 + threadIdx.x], lsq[threadIdx.x]);
    }
}

// fold BN into affine: s = g*rsqrt(var+eps), t = beta - mu*s   (16 threads)
__global__ void k_bnp(const float* __restrict__ sums, const float* __restrict__ gamma,
                      const float* __restrict__ beta, float* __restrict__ s,
                      float* __restrict__ t, float cnt) {
    int f = threadIdx.x;
    if (f < 16) {
        float mu = sums[f] / cnt;
        float var = sums[16 + f] / cnt - mu * mu;
        float sc = gamma[f] * rsqrtf(var + BN_EPS);
        s[f] = sc;
        t[f] = beta[f] - mu * sc;
    }
}

// ---------------------------------------------------------------------------
// GEMM2: xs2[n,64] = ((h1*s1+t1) @ W2) * rsqrt(max(deg_out,1))  (K=16 pad 32)
// ---------------------------------------------------------------------------
__global__ void k_gemm2(const float* __restrict__ h1, const float* __restrict__ W2,
                        const float* __restrict__ s1, const float* __restrict__ t1,
                        const float* __restrict__ deg_out, float* __restrict__ xs2, int N) {
    __shared__ v16h lB[4 * 32];                 // [nt][lane], K padded 16->32
    __shared__ float ls[16], lt[16];
    {
        _Float16* pb = (_Float16*)lB;
        for (int idx = threadIdx.x; idx < 4 * 32 * 16; idx += blockDim.x) {
            int nt = idx >> 9, L = (idx >> 4) & 31, i = idx & 15;
            int k = ((L >> 4) << 4) + i;
            pb[idx] = (k < 16) ? (_Float16)W2[k * 64 + nt * 16 + (L & 15)] : (_Float16)0.0f;
        }
    }
    if (threadIdx.x < 16) { ls[threadIdx.x] = s1[threadIdx.x]; lt[threadIdx.x] = t1[threadIdx.x]; }
    __syncthreads();

    int wv = threadIdx.x >> 5, lane = threadIdx.x & 31;
    int tile = blockIdx.x * (blockDim.x >> 5) + wv;
    int m0 = tile * 16;
    if (m0 >= N) return;
    int h = lane >> 4, col = lane & 15;
    int row = m0 + (lane & 15); if (row >= N) row = N - 1;
    const float* hr = h1 + (size_t)row * 16;

    v16h a;
#pragma unroll
    for (int i = 0; i < 16; ++i) {
        int k = a16_k(i, h);
        a[i] = (k < 16) ? (_Float16)(hr[k] * ls[k] + lt[k]) : (_Float16)0.0f;
    }
#pragma unroll
    for (int nt = 0; nt < 4; ++nt) {
        v8f c = {};
        c = wmma_f16(a, lB[nt * 32 + lane], c);
#pragma unroll
        for (int v = 0; v < 8; ++v) {
            int r = m0 + v + (h << 3);
            if (r < N) {
                float rs = rsqrtf(fmaxf(deg_out[r], 1.0f));
                xs2[(size_t)r * 64 + nt * 16 + col] = c[v] * rs;
            }
        }
    }
}

// finalize GC2 -> f16 embeddings for the edge GEMM gathers
__global__ void k_fin2(const float* __restrict__ agg2, const float* __restrict__ deg_in,
                       const float* __restrict__ b2, _Float16* __restrict__ h2h, int N) {
    long long idx = (long long)blockIdx.x * blockDim.x + threadIdx.x;
    if (idx < (long long)N * 64) {
        int nd = (int)(idx >> 6), f = (int)(idx & 63);
        float v = agg2[idx] * rsqrtf(fmaxf(deg_in[nd], 1.0f)) + b2[f];
        h2h[idx] = (_Float16)v;
    }
}

// ---------------------------------------------------------------------------
// Edge GEMM: z[e,16] = concat(e_feats, h2[src], h2[dst]) @ Wp1 + bp1 (K=160)
// ---------------------------------------------------------------------------
__global__ void k_egemm(const float* __restrict__ ef, const _Float16* __restrict__ h2h,
                        const int* __restrict__ src, const int* __restrict__ dst,
                        const float* __restrict__ Wp1, const float* __restrict__ bp1,
                        float* __restrict__ z, float* __restrict__ stats, int E) {
    __shared__ v16h lB[5 * 32];                 // [kk][lane], pre-swizzled fragments
    __shared__ float lsum[16], lsq[16];
    {
        _Float16* pb = (_Float16*)lB;
        for (int idx = threadIdx.x; idx < 5 * 32 * 16; idx += blockDim.x) {
            int kk = idx >> 9, L = (idx >> 4) & 31, i = idx & 15;
            int k = kk * 32 + ((L >> 4) << 4) + i;
            pb[idx] = (_Float16)Wp1[k * 16 + (L & 15)];
        }
    }
    if (threadIdx.x < 16) { lsum[threadIdx.x] = 0.f; lsq[threadIdx.x] = 0.f; }
    __syncthreads();

    int wv = threadIdx.x >> 5, lane = threadIdx.x & 31;
    int tile = blockIdx.x * (blockDim.x >> 5) + wv;
    int h = lane >> 4, col = lane & 15;

    if (tile * 16 < E) {
        int e = tile * 16 + (lane & 15); if (e >= E) e = E - 1;
        int s = src[e], d = dst[e];
        const float*    er = ef  + (size_t)e * 32;
        const _Float16* sr = h2h + (size_t)s * 64;
        const _Float16* dr = h2h + (size_t)d * 64;

        v8f c = {};
        { // K 0..31 : edge features (f32 -> f16)
            v16h a;
#pragma unroll
            for (int i = 0; i < 16; ++i) a[i] = (_Float16)er[a16_k(i, h)];
            c = wmma_f16(a, lB[0 * 32 + lane], c);
        }
#pragma unroll
        for (int kk = 0; kk < 2; ++kk) { // K 32..95 : h2[src]
            v16h a;
#pragma unroll
            for (int i = 0; i < 16; ++i) a[i] = sr[kk * 32 + a16_k(i, h)];
            c = wmma_f16(a, lB[(1 + kk) * 32 + lane], c);
        }
#pragma unroll
        for (int kk = 0; kk < 2; ++kk) { // K 96..159 : h2[dst]
            v16h a;
#pragma unroll
            for (int i = 0; i < 16; ++i) a[i] = dr[kk * 32 + a16_k(i, h)];
            c = wmma_f16(a, lB[(3 + kk) * 32 + lane], c);
        }

        float bn = bp1[col];
        float psum = 0.f, psq = 0.f;
#pragma unroll
        for (int v = 0; v < 8; ++v) {
            int r = tile * 16 + v + (h << 3);
            if (r < E) {
                float val = c[v] + bn;
                z[(size_t)r * 16 + col] = val;
                psum += val; psq += val * val;
            }
        }
        atomicAdd(&lsum[col], psum);
        atomicAdd(&lsq[col], psq);
    }
    __syncthreads();
    if (threadIdx.x < 16) {
        atomicAdd(&stats[32 + threadIdx.x], lsum[threadIdx.x]);
        atomicAdd(&stats[48 + threadIdx.x], lsq[threadIdx.x]);
    }
}

// ---------------------------------------------------------------------------
// Head: BN affine -> Linear(16->8) -> log_softmax
// ---------------------------------------------------------------------------
__global__ void k_head(const float* __restrict__ z, const float* __restrict__ sp,
                       const float* __restrict__ tp, const float* __restrict__ Wp2,
                       const float* __restrict__ bp2, float* __restrict__ out, int E) {
    __shared__ float lW[16 * 8], lb[8], ls[16], lt[16];
    if (threadIdx.x < 128) lW[threadIdx.x] = Wp2[threadIdx.x];
    if (threadIdx.x < 8)   lb[threadIdx.x] = bp2[threadIdx.x];
    if (threadIdx.x < 16)  { ls[threadIdx.x] = sp[threadIdx.x]; lt[threadIdx.x] = tp[threadIdx.x]; }
    __syncthreads();
    int e = blockIdx.x * blockDim.x + threadIdx.x;
    if (e >= E) return;
    const float* zr = z + (size_t)e * 16;
    float zb[16];
#pragma unroll
    for (int f = 0; f < 16; ++f) zb[f] = zr[f] * ls[f] + lt[f];
    float l[8];
#pragma unroll
    for (int cc = 0; cc < 8; ++cc) {
        float acc = lb[cc];
#pragma unroll
        for (int f = 0; f < 16; ++f) acc = fmaf(zb[f], lW[f * 8 + cc], acc);
        l[cc] = acc;
    }
    float m = l[0];
#pragma unroll
    for (int cc = 1; cc < 8; ++cc) m = fmaxf(m, l[cc]);
    float se = 0.f;
#pragma unroll
    for (int cc = 0; cc < 8; ++cc) se += expf(l[cc] - m);
    float lse = m + logf(se);
#pragma unroll
    for (int cc = 0; cc < 8; ++cc) out[(size_t)e * 8 + cc] = l[cc] - lse;
}

// ---------------------------------------------------------------------------
extern "C" void kernel_launch(void* const* d_in, const int* in_sizes, int n_in,
                              void* d_out, int out_size, void* d_ws, size_t ws_size,
                              hipStream_t stream) {
    const float* nf  = (const float*)d_in[0];
    const float* ef  = (const float*)d_in[1];
    const int*   src = (const int*)d_in[2];
    const int*   dst = (const int*)d_in[3];
    const float* W1  = (const float*)d_in[4];
    const float* b1  = (const float*)d_in[5];
    const float* g1  = (const float*)d_in[6];
    const float* be1 = (const float*)d_in[7];
    const float* W2  = (const float*)d_in[8];
    const float* b2  = (const float*)d_in[9];
    const float* Wp1 = (const float*)d_in[10];
    const float* bp1 = (const float*)d_in[11];
    const float* gp  = (const float*)d_in[12];
    const float* bep = (const float*)d_in[13];
    const float* Wp2 = (const float*)d_in[14];
    const float* bp2 = (const float*)d_in[15];

    const int N = in_sizes[0] / 128;
    const int E = in_sizes[2];

    float* ws = (float*)d_ws;
    size_t o = 0;
    auto take = [&](size_t cnt) { size_t r = o; o += (cnt + 15) & ~(size_t)15; return r; };
    // atomically-accumulated regions first (zeroed each call)
    size_t o_degout = take((size_t)N);
    size_t o_degin  = take((size_t)N);
    size_t o_agg1   = take((size_t)N * 16);
    size_t o_agg2   = take((size_t)N * 64);
    size_t o_stats  = take(128);
    long long zeroN = (long long)o;
    // overwrite-before-read regions
    size_t o_xs1 = take((size_t)N * 16);   // reused as h1 after scatter1
    size_t o_xs2 = take((size_t)N * 64);
    size_t o_h2h = take((size_t)N * 32);   // f16 [N,64]
    size_t o_z   = take((size_t)E * 16);

    float* stats = ws + o_stats;           // [0:16) h1sum [16:32) h1sq [32:48) zsum [48:64) zsq
    float* s1 = stats + 64; float* t1 = stats + 80;
    float* sp = stats + 96; float* tp = stats + 112;

    const int TB = 256;
    const int tilesN = (N + 15) / 16;
    const int tilesE = (E + 15) / 16;

    k_zero<<<(unsigned)((zeroN + TB - 1) / TB), TB, 0, stream>>>(ws, zeroN);
    k_deg<<<(E + TB - 1) / TB, TB, 0, stream>>>(src, dst, ws + o_degout, ws + o_degin, E);
    k_gemm1<<<(tilesN + 7) / 8, 256, 0, stream>>>(nf, W1, ws + o_degout, ws + o_xs1, N);
    k_scat<16><<<(unsigned)(((long long)E * 16 + TB - 1) / TB), TB, 0, stream>>>(
        src, dst, ws + o_xs1, ws + o_agg1, E);
    k_fin1<<<(unsigned)(((long long)N * 16 + TB - 1) / TB), TB, 0, stream>>>(
        ws + o_agg1, ws + o_degin, b1, ws + o_xs1, stats, N);
    k_bnp<<<1, 16, 0, stream>>>(stats, g1, be1, s1, t1, (float)N);
    k_gemm2<<<(tilesN + 7) / 8, 256, 0, stream>>>(ws + o_xs1, W2, s1, t1,
                                                  ws + o_degout, ws + o_xs2, N);
    k_scat<64><<<(unsigned)(((long long)E * 64 + TB - 1) / TB), TB, 0, stream>>>(
        src, dst, ws + o_xs2, ws + o_agg2, E);
    k_fin2<<<(unsigned)(((long long)N * 64 + TB - 1) / TB), TB, 0, stream>>>(
        ws + o_agg2, ws + o_degin, b2, (_Float16*)(ws + o_h2h), N);
    k_egemm<<<(tilesE + 7) / 8, 256, 0, stream>>>(ef, (const _Float16*)(ws + o_h2h),
                                                  src, dst, Wp1, bp1, ws + o_z, stats, E);
    k_bnp<<<1, 16, 0, stream>>>(stats + 32, gp, bep, sp, tp, (float)E);
    k_head<<<(E + TB - 1) / TB, TB, 0, stream>>>(ws + o_z, sp, tp, Wp2, bp2,
                                                 (float*)d_out, E);
}
